// MultiHeadAttentionQuantum_65481071402350
// MI455X (gfx1250) — matrix operand order repaired and checked by reading, not compiled, not run on hardware
//
#include <hip/hip_runtime.h>
#include <hip/hip_bf16.h>

typedef __attribute__((ext_vector_type(16))) _Float16 v16h;
typedef __attribute__((ext_vector_type(8)))  float    v8f;

#define NQ    8
#define SEQ   2048
#define ED    8
#define BATCH 8

struct c2 { float x, y; };
__device__ __forceinline__ c2 cmul(c2 a, c2 b) {
    return { a.x*b.x - a.y*b.y, a.x*b.y + a.y*b.x };
}

// ---------------------------------------------------------------------------
// Kernel 1: quantum head. One wave32 per row; 256-amp state held as 8 complex
// amplitudes per lane (j = (lane<<3)|r). CNOT chain == permutation
// y -> y^(y>>1): cross-lane part is a single uniform __shfl from lane^(lane>>1),
// in-lane part is a static gray-code register permute. H hits bit0 (in-lane).
// ---------------------------------------------------------------------------
__global__ void __launch_bounds__(256) qhead_kernel(const float* __restrict__ x,
                                                    _Float16* __restrict__ qkv)
{
    const int lane = threadIdx.x & 31;
    const int row  = (blockIdx.x * blockDim.x + threadIdx.x) >> 5;

    float c[NQ], s[NQ];
#pragma unroll
    for (int w = 0; w < NQ; ++w) {
        float a = 0.5f * x[row * NQ + w];
        c[w] = __cosf(a);
        s[w] = -__sinf(a);              // RX |0> amplitude for |1> is -i*sin -> (0, s)
    }

    // wires 0..4 <-> index bits 7..3 <-> lane bits 4..0
    c2 base = {1.f, 0.f};
#pragma unroll
    for (int w = 0; w < 5; ++w) {
        int bit = (lane >> (4 - w)) & 1;
        c2 f = bit ? c2{0.f, s[w]} : c2{c[w], 0.f};
        base = cmul(base, f);
    }
    c2 amp[8];
#pragma unroll
    for (int r = 0; r < 8; ++r) {       // wires 5,6,7 <-> r bits 2,1,0
        c2 t = base;
        t = cmul(t, (r & 4) ? c2{0.f, s[5]} : c2{c[5], 0.f});
        t = cmul(t, (r & 2) ? c2{0.f, s[6]} : c2{c[6], 0.f});
        t = cmul(t, (r & 1) ? c2{0.f, s[7]} : c2{c[7], 0.f});
        amp[r] = t;
    }

    const float inv_s2  = 0.70710678118654752f;
    const int   srcLane = lane ^ (lane >> 1);
    const bool  par     = lane & 1;
#pragma unroll
    for (int layer = 0; layer < 2; ++layer) {
        c2 sh[8];
#pragma unroll
        for (int r = 0; r < 8; ++r) {
            sh[r].x = __shfl(amp[r].x, srcLane, 32);
            sh[r].y = __shfl(amp[r].y, srcLane, 32);
        }
        c2 g[8];
#pragma unroll
        for (int r = 0; r < 8; ++r) {
            int gr = r ^ (r >> 1);
            g[r] = par ? sh[gr ^ 4] : sh[gr];
        }
#pragma unroll
        for (int k = 0; k < 4; ++k) {   // H on last wire (index bit 0)
            c2 e = g[2*k], o = g[2*k + 1];
            amp[2*k]     = { (e.x + o.x)*inv_s2, (e.y + o.y)*inv_s2 };
            amp[2*k + 1] = { (e.x - o.x)*inv_s2, (e.y - o.y)*inv_s2 };
        }
    }

    float p[8];
#pragma unroll
    for (int r = 0; r < 8; ++r) p[r] = amp[r].x*amp[r].x + amp[r].y*amp[r].y;

    float t[8];
    float tot = 0.f;
#pragma unroll
    for (int r = 0; r < 8; ++r) tot += p[r];
#pragma unroll
    for (int w = 0; w < 5; ++w)         // sign from lane bit (4-w)
        t[w] = ((lane >> (4 - w)) & 1) ? -tot : tot;
    t[5] = t[6] = t[7] = 0.f;
#pragma unroll
    for (int r = 0; r < 8; ++r) {
        t[5] += (r & 4) ? -p[r] : p[r];
        t[6] += (r & 2) ? -p[r] : p[r];
        t[7] += (r & 1) ? -p[r] : p[r];
    }
#pragma unroll
    for (int w = 0; w < 8; ++w) {
#pragma unroll
        for (int m = 1; m < 32; m <<= 1) t[w] += __shfl_xor(t[w], m, 32);
    }
    if (lane == 0) {
        union { _Float16 h[8]; uint4 u; } pk;
#pragma unroll
        for (int w = 0; w < 8; ++w) pk.h[w] = (_Float16)t[w];
        *reinterpret_cast<uint4*>(qkv + row * 8) = pk.u;   // 16B aligned
    }
}

// ---------------------------------------------------------------------------
// Kernel 2: flash attention, one wave per 16-query tile, 8 waves/block.
// Streams keys 32 at a time: 2x v_wmma_f32_16x16x32_f16 for S (K padded 8->32),
// online softmax in C-layout with shfl_xor row reductions, P staged via LDS
// to A-layout, 1x WMMA for P@V (K=32). Epilogue applies W_out^T + b_out.
// ---------------------------------------------------------------------------
__global__ void __launch_bounds__(256) attn_kernel(const _Float16* __restrict__ qkv,
                                                   const int*      __restrict__ mask,
                                                   const float*    __restrict__ Wout,
                                                   const float*    __restrict__ bout,
                                                   float*          __restrict__ out)
{
    __shared__ _Float16 pstage[8][16 * 32];   // per-wave P tile (half)
    __shared__ float    ostage[8][16 * 8];    // per-wave O tile

    const int lane = threadIdx.x & 31;
    const int wv   = threadIdx.x >> 5;
    const int tile = blockIdx.x * 8 + wv;     // 0..1023
    const int b    = tile >> 7;               // 128 q-tiles per batch
    const int qt   = tile & 127;
    const int hi   = lane >> 4;               // C-layout row-half
    const int nn   = lane & 15;               // C-layout column / B-layout N

    const _Float16* qb = qkv + (size_t)b * SEQ * ED;
    const int q0 = qt * 16;
    const float scale = 0.35355339059327373f; // 1/sqrt(8) folded into Q

    // A operand: Q tile 16x32f16.  Lanes 0-15 h[0..7] = K 0..7 (d<8), rest zero.
    v16h aq;
#pragma unroll
    for (int i = 0; i < 16; ++i) aq[i] = (_Float16)0.0f;
    if (hi == 0) {
#pragma unroll
        for (int d = 0; d < 8; ++d)
            aq[d] = (_Float16)((float)qb[(q0 + nn) * ED + d] * scale);
    }

    float m_i[8], l_i[8];
#pragma unroll
    for (int r = 0; r < 8; ++r) { m_i[r] = -INFINITY; l_i[r] = 0.f; }
    v8f acc = {};

    const size_t mrow0 = ((size_t)b * SEQ + q0) * SEQ;

    for (int kc = 0; kc < SEQ; kc += 32) {
        // B operands: K^T for keys kc..kc+15 / kc+16..kc+31.
        // B layout: lane l -> n=l&15, h[i] = B[K][n], K = 8*(l>>4)+i (i<8) / 16+8*(l>>4)+i-8; K>=8 rows zero.
        v16h bk0, bk1;
#pragma unroll
        for (int i = 0; i < 16; ++i) { bk0[i] = (_Float16)0.0f; bk1[i] = (_Float16)0.0f; }
        if (hi == 0) {
#pragma unroll
            for (int d = 0; d < 8; ++d) {
                bk0[d] = qb[(kc + nn) * ED + d];
                bk1[d] = qb[(kc + 16 + nn) * ED + d];
            }
        }
        v8f z0 = {}, z1 = {};
        v8f s0 = __builtin_amdgcn_wmma_f32_16x16x32_f16(false, aq, false, bk0,
                                                        (short)0, z0, false, false);
        v8f s1 = __builtin_amdgcn_wmma_f32_16x16x32_f16(false, aq, false, bk1,
                                                        (short)0, z1, false, false);

        // mask (element (m = r+8*hi, key = kc[+16]+nn) lives in this lane's C slot)
#pragma unroll
        for (int r = 0; r < 8; ++r) {
            size_t mb = mrow0 + (size_t)(r + 8 * hi) * SEQ;
            if (mask[mb + kc + nn] == 0)      s0[r] = -1.0e9f;
            if (mask[mb + kc + 16 + nn] == 0) s1[r] = -1.0e9f;
        }

        // online softmax (row = 16 lanes of this half; xor masks <16 stay in half)
        float mnew[8], alpha[8];
#pragma unroll
        for (int r = 0; r < 8; ++r) {
            float v = fmaxf(s0[r], s1[r]);
            v = fmaxf(v, __shfl_xor(v, 1, 32));
            v = fmaxf(v, __shfl_xor(v, 2, 32));
            v = fmaxf(v, __shfl_xor(v, 4, 32));
            v = fmaxf(v, __shfl_xor(v, 8, 32));
            mnew[r]  = fmaxf(m_i[r], v);
            alpha[r] = __expf(m_i[r] - mnew[r]);
            m_i[r]   = mnew[r];
        }
        float p0[8], p1[8];
#pragma unroll
        for (int r = 0; r < 8; ++r) {
            p0[r] = __expf(s0[r] - mnew[r]);
            p1[r] = __expf(s1[r] - mnew[r]);
            float rs = p0[r] + p1[r];
            rs += __shfl_xor(rs, 1, 32);
            rs += __shfl_xor(rs, 2, 32);
            rs += __shfl_xor(rs, 4, 32);
            rs += __shfl_xor(rs, 8, 32);
            l_i[r] = l_i[r] * alpha[r] + rs;
            acc[r] *= alpha[r];
        }

        // stage P (C-layout -> LDS row-major 16x32 half)
#pragma unroll
        for (int r = 0; r < 8; ++r) {
            int m = r + 8 * hi;
            pstage[wv][m * 32 + nn]      = (_Float16)p0[r];
            pstage[wv][m * 32 + 16 + nn] = (_Float16)p1[r];
        }
        asm volatile("s_wait_dscnt 0" ::: "memory");   // same-wave LDS RAW

        // A operand: P 16x32.  lane: m=nn, h[0..7]=K hi*8.., h[8..15]=K 16+hi*8..
        v16h ap;
        {
            const _Float16* pr = &pstage[wv][nn * 32];
#pragma unroll
            for (int i = 0; i < 8; ++i) ap[i]     = pr[hi * 8 + i];
#pragma unroll
            for (int i = 0; i < 8; ++i) ap[8 + i] = pr[16 + hi * 8 + i];
        }
        // B operand: V chunk 32x16 (cols 8..15 zero)
        v16h bv;
#pragma unroll
        for (int i = 0; i < 16; ++i) bv[i] = (_Float16)0.0f;
        if (nn < 8) {
#pragma unroll
            for (int i = 0; i < 16; ++i)
                bv[i] = qb[(kc + 8 * hi + i + (i >= 8 ? 8 : 0)) * ED + nn];
        }
        acc = __builtin_amdgcn_wmma_f32_16x16x32_f16(false, ap, false, bv,
                                                     (short)0, acc, false, false);
    }

    // normalize, stage O
    if (nn < 8) {
#pragma unroll
        for (int r = 0; r < 8; ++r)
            ostage[wv][(r + 8 * hi) * 8 + nn] = acc[r] / l_i[r];
    }
    asm volatile("s_wait_dscnt 0" ::: "memory");

    // epilogue: out[m][e] = b[e] + sum_d O[m][d] * Wout[e][d];  4 outputs/lane
#pragma unroll
    for (int t = 0; t < 4; ++t) {
        int oi = lane + 32 * t;          // 0..127
        int m = oi >> 3, e = oi & 7;
        float v = bout[e];
#pragma unroll
        for (int d = 0; d < 8; ++d)
            v += ostage[wv][m * 8 + d] * Wout[e * 8 + d];
        out[((size_t)b * SEQ + q0 + m) * ED + e] = v;
    }
}

extern "C" void kernel_launch(void* const* d_in, const int* in_sizes, int n_in,
                              void* d_out, int out_size, void* d_ws, size_t ws_size,
                              hipStream_t stream)
{
    const float* x    = (const float*)d_in[0];   // (8,2048,8) f32
    const int*   mask = (const int*)  d_in[1];   // (8,2048,2048) i32
    const float* Wout = (const float*)d_in[2];   // (8,8) f32
    const float* bout = (const float*)d_in[3];   // (8,) f32
    float* out = (float*)d_out;                  // (8,2048,8) f32

    _Float16* qkv = (_Float16*)d_ws;             // 16384*8 halves = 256 KB

    // 16384 rows, one wave each, 8 waves/block -> 2048 blocks
    qhead_kernel<<<2048, 256, 0, stream>>>(x, qkv);
    // 1024 query tiles (8 batches x 128), one wave each, 8 waves/block
    attn_kernel<<<128, 256, 0, stream>>>(qkv, mask, Wout, bout, out);
}
